// Corner_pooling_module_52201032515724
// MI455X (gfx1250) — matrix operand order repaired and checked by reading, not compiled, not run on hardware
//
#include <hip/hip_runtime.h>
#include <stdint.h>
#include <stddef.h>

// ---------------------------------------------------------------------------
// Types
// ---------------------------------------------------------------------------
typedef __bf16 bf16_t;
typedef __attribute__((ext_vector_type(16))) __bf16 v16bf;
typedef __attribute__((ext_vector_type(8)))  __bf16 v8bf;
typedef __attribute__((ext_vector_type(8)))  float  v8f;

static constexpr int  IMG_H  = 128;
static constexpr int  IMG_W  = 128;
static constexpr int  HWX    = IMG_H * IMG_W;   // 16384
static constexpr int  NBATCH = 4;

static __device__ __forceinline__ float  bf2f(bf16_t b) { return (float)b; }
static __device__ __forceinline__ bf16_t f2bf(float f)  { return (bf16_t)f; }

union Frag { v16bf v; v8bf h[2]; };

static __device__ __forceinline__ v8bf zero8() {
  v8bf z;
#pragma unroll
  for (int i = 0; i < 8; ++i) z[i] = (bf16_t)0.0f;
  return z;
}

// ---------------------------------------------------------------------------
// Implicit-GEMM conv (SAME pad), bf16 WMMA, fused BN (+residual) (+ReLU).
//   in  : NHWC bf16  (NBATCH, H, W, CIN)
//   wp  : packed weights bf16 [COUT][K], K = KDIM*KDIM*CIN, k = tap*CIN + cin
//   res : optional residual, NCHW f32  (only when ADD_RES)
//   out : NHWC bf16 (OUT_NCHW=false) or NCHW f32 (OUT_NCHW=true)
//
// Block = 256 threads = 8 waves, arranged 2(M) x 4(N).
// Wave tile = 32 (M) x 16*NTJ (N): 2 x NTJ WMMA frags -> 2*NTJ WMMAs per
// 32-deep k-chunk on (4 + 2*NTJ) 16B fragment loads.
// Block tile = 64 (spatial, one row strip) x 64*NTJ (channels).
// ---------------------------------------------------------------------------
template <int CIN, int COUT, int KDIM, int NTJ, bool RELU, bool ADD_RES, bool OUT_NCHW>
__global__ __launch_bounds__(256)
void conv_wmma_kernel(const bf16_t* __restrict__ in,
                      const bf16_t* __restrict__ wp,
                      const float*  __restrict__ sc,
                      const float*  __restrict__ bi,
                      const float*  __restrict__ res,
                      bf16_t*       __restrict__ outb,
                      float*        __restrict__ outf) {
  constexpr int K   = KDIM * KDIM * CIN;
  constexpr int PAD = KDIM / 2;

  const int b    = blockIdx.z;
  const int m0   = blockIdx.x * 64;         // flat spatial start (one row strip)
  const int h    = m0 >> 7;                 // / IMG_W
  const int w0   = m0 & (IMG_W - 1);
  const int nblk = blockIdx.y * (64 * NTJ);

  const int lane = threadIdx.x & 31;
  const int wid  = threadIdx.x >> 5;
  const int wr   = wid >> 2;                // 0..1  (M)
  const int wc   = wid & 3;                 // 0..3  (N)
  const int half = lane >> 4;               // 0/1
  const int l15  = lane & 15;

  // per-lane spatial w for the two M sub-tiles (A rows)
  const int wA[2] = { w0 + wr * 32 + l15, w0 + wr * 32 + 16 + l15 };
  // N sub-tile bases (NTJ tiles of 16 per wave)
  int nT[NTJ];
#pragma unroll
  for (int j = 0; j < NTJ; ++j) nT[j] = nblk + wc * (16 * NTJ) + j * 16;

  v8f acc[2][NTJ] = {};

  const size_t img = (size_t)b * HWX;

#pragma unroll
  for (int kh = 0; kh < KDIM; ++kh) {
    const int  hh  = h + kh - PAD;
    const bool hok = (unsigned)hh < (unsigned)IMG_H;
#pragma unroll
    for (int kw = 0; kw < KDIM; ++kw) {
      const int tap = kh * KDIM + kw;
      const bf16_t* apk[2];
      bool av[2];
#pragma unroll
      for (int i = 0; i < 2; ++i) {
        const int ww = wA[i] + kw - PAD;
        av[i]  = hok && ((unsigned)ww < (unsigned)IMG_W);
        apk[i] = in + (img + (size_t)hh * IMG_W + ww) * CIN;
      }
      const size_t kofs = (size_t)tap * CIN;

      // one speculative prefetch per tap: next tap's weights for this lane
      if (tap + 1 < KDIM * KDIM)
        __builtin_prefetch(wp + (size_t)(nT[0] + l15) * K + kofs + CIN, 0, 3);

#pragma unroll
      for (int c0 = 0; c0 < CIN; c0 += 32) {
        // --- A fragments: 16x32 bf16. lane<16: rows 0..15, k {0..7,16..23};
        //     lane>=16: same rows, k {8..15,24..31}. Two contiguous 16B chunks.
        Frag a[2];
#pragma unroll
        for (int i = 0; i < 2; ++i) {
          if (av[i]) {
            const bf16_t* p = apk[i] + c0 + half * 8;
            a[i].h[0] = *(const v8bf*)(p);
            a[i].h[1] = *(const v8bf*)(p + 16);
          } else {
            a[i].h[0] = zero8();
            a[i].h[1] = zero8();
          }
        }
        // --- B fragments: 32x16 bf16. lane<16: k {0..7,16..23} of column l15;
        //     lane>=16: k {8..15,24..31}. Two contiguous 16B chunks.
        Frag bm[NTJ];
#pragma unroll
        for (int j = 0; j < NTJ; ++j) {
          const bf16_t* p = wp + (size_t)(nT[j] + l15) * K + kofs + c0 + half * 8;
          bm[j].h[0] = *(const v8bf*)(p);
          bm[j].h[1] = *(const v8bf*)(p + 16);
        }
#pragma unroll
        for (int i = 0; i < 2; ++i)
#pragma unroll
          for (int j = 0; j < NTJ; ++j)
            acc[i][j] = __builtin_amdgcn_wmma_f32_16x16x32_bf16(
                false, a[i].v, false, bm[j].v, (short)0, acc[i][j], false, false);
      }
    }
  }

  // Epilogue: D layout (32-bit C/D 16x16): lane<16 -> M=r, N=lane;
  //           lane>=16 -> M=8+r, N=lane-16.  M == w offset within strip.
#pragma unroll
  for (int j = 0; j < NTJ; ++j) {
    const int   n  = nT[j] + l15;
    const float s_ = sc[n];
    const float b_ = bi[n];
#pragma unroll
    for (int i = 0; i < 2; ++i) {
      const int wbase = w0 + wr * 32 + i * 16 + half * 8;
#pragma unroll
      for (int r = 0; r < 8; ++r) {
        const int wq = wbase + r;
        float v = acc[i][j][r] * s_ + b_;
        if constexpr (ADD_RES)
          v += res[((size_t)b * COUT + n) * HWX + (size_t)h * IMG_W + wq];
        if constexpr (RELU) v = fmaxf(v, 0.0f);
        if constexpr (OUT_NCHW)
          outf[((size_t)b * COUT + n) * HWX + (size_t)h * IMG_W + wq] = v;
        else
          outb[(img + (size_t)h * IMG_W + wq) * COUT + n] = f2bf(v);
      }
    }
  }
}

// ---------------------------------------------------------------------------
// Reverse cummax over H, in place, NHWC bf16 with C=128. Thread per (b,w,c):
// coalesced across (w,c) at each h step.
// ---------------------------------------------------------------------------
__global__ __launch_bounds__(256)
void scan_h_kernel(bf16_t* __restrict__ p) {
  const int tid = blockIdx.x * 256 + threadIdx.x;   // 4*128*128 threads
  const int c = tid & 127;
  const int w = (tid >> 7) & 127;
  const int b = tid >> 14;
  size_t idx = (((size_t)b * HWX) + (size_t)(IMG_H - 1) * IMG_W + w) * 128 + c;
  float m = bf2f(p[idx]);
  for (int h = IMG_H - 2; h >= 0; --h) {
    idx -= (size_t)IMG_W * 128;
    m = fmaxf(m, bf2f(p[idx]));
    p[idx] = f2bf(m);
  }
}

// ---------------------------------------------------------------------------
// Reverse cummax over W on p2 (in place) fused with s = cp2 + cp1.
// Thread per (b,h,c).
// ---------------------------------------------------------------------------
__global__ __launch_bounds__(256)
void scan_w_add_kernel(bf16_t* __restrict__ p2, const bf16_t* __restrict__ cp1) {
  const int tid = blockIdx.x * 256 + threadIdx.x;   // 4*128*128 threads
  const int c = tid & 127;
  const int h = (tid >> 7) & 127;
  const int b = tid >> 14;
  size_t idx = (((size_t)b * HWX) + (size_t)h * IMG_W + (IMG_W - 1)) * 128 + c;
  float m = bf2f(p2[idx]);
  p2[idx] = f2bf(m + bf2f(cp1[idx]));
  for (int w = IMG_W - 2; w >= 0; --w) {
    idx -= 128;
    m = fmaxf(m, bf2f(p2[idx]));
    p2[idx] = f2bf(m + bf2f(cp1[idx]));
  }
}

// ---------------------------------------------------------------------------
// NCHW f32 -> NHWC bf16, LDS-tiled 32x32 transpose (C=256).
// grid: (HWX/32, 256/32, NBATCH), block 256.
// ---------------------------------------------------------------------------
__global__ __launch_bounds__(256)
void cvt_nchw_to_nhwc_kernel(const float* __restrict__ x, bf16_t* __restrict__ xbf) {
  __shared__ float tile[32][33];
  const int t   = threadIdx.x;
  const int b   = blockIdx.z;
  const int c0  = blockIdx.y * 32;
  const int hw0 = blockIdx.x * 32;
  const int r   = t >> 5;     // 0..7
  const int col = t & 31;
#pragma unroll
  for (int it = 0; it < 4; ++it) {
    const int cc = r + it * 8;
    tile[cc][col] = x[((size_t)b * 256 + c0 + cc) * HWX + hw0 + col];
  }
  __syncthreads();
#pragma unroll
  for (int it = 0; it < 4; ++it) {
    const int hwl = r + it * 8;
    xbf[((size_t)b * HWX + hw0 + hwl) * 256 + c0 + col] = f2bf(tile[col][hwl]);
  }
}

// ---------------------------------------------------------------------------
// Weight repack: OIHW f32 -> [O][K] bf16 with k = tap*Cin + cin.
// ---------------------------------------------------------------------------
__global__ __launch_bounds__(256)
void pack_w_kernel(const float* __restrict__ w, bf16_t* __restrict__ wp,
                   int Cout, int Cin, int KD) {
  const int K     = KD * KD * Cin;
  const int total = Cout * K;
  const int tid   = blockIdx.x * 256 + threadIdx.x;
  if (tid >= total) return;
  const int n   = tid / K;
  const int k   = tid - n * K;
  const int tap = k / Cin;
  const int cin = k - tap * Cin;
  wp[tid] = f2bf(w[((size_t)n * Cin + cin) * (KD * KD) + tap]);
}

// ---------------------------------------------------------------------------
// BN fold: sc = g * rsqrt(v + eps), bi = b - m * sc
// ---------------------------------------------------------------------------
__global__ __launch_bounds__(256)
void bn_prep_kernel(const float* __restrict__ g, const float* __restrict__ b,
                    const float* __restrict__ m, const float* __restrict__ v,
                    float* __restrict__ sc, float* __restrict__ bi, int C) {
  const int i = blockIdx.x * 256 + threadIdx.x;
  if (i < C) {
    const float s = g[i] * rsqrtf(v[i] + 1e-5f);
    sc[i] = s;
    bi[i] = b[i] - m[i] * s;
  }
}

// ---------------------------------------------------------------------------
// Host launcher
// ---------------------------------------------------------------------------
extern "C" void kernel_launch(void* const* d_in, const int* in_sizes, int n_in,
                              void* d_out, int out_size, void* d_ws, size_t ws_size,
                              hipStream_t stream) {
  (void)in_sizes; (void)n_in; (void)out_size; (void)ws_size;

  const float* x    = (const float*)d_in[0];
  const float* w_p1 = (const float*)d_in[1];
  const float* w_p2 = (const float*)d_in[6];
  const float* w_c1 = (const float*)d_in[11];
  const float* w_c2 = (const float*)d_in[16];
  const float* w_p3 = (const float*)d_in[21];
  float* out = (float*)d_out;

  // ---- workspace carve-up (256B aligned) ----
  char*  ws  = (char*)d_ws;
  size_t off = 0;
  auto take = [&](size_t bytes) -> void* {
    void* p = ws + off;
    off += (bytes + 255) & ~(size_t)255;
    return p;
  };
  bf16_t* xbf = (bf16_t*)take((size_t)NBATCH * HWX * 256 * 2);   // NHWC x (33.5MB)
  bf16_t* p1  = (bf16_t*)take((size_t)NBATCH * HWX * 128 * 2);   // 16.8MB
  bf16_t* p2  = (bf16_t*)take((size_t)NBATCH * HWX * 128 * 2);   // 16.8MB (becomes s)
  bf16_t* wp1 = (bf16_t*)take((size_t)128 * 2304 * 2);
  bf16_t* wp2 = (bf16_t*)take((size_t)128 * 2304 * 2);
  bf16_t* wc1 = (bf16_t*)take((size_t)256 * 1152 * 2);
  bf16_t* wc2 = (bf16_t*)take((size_t)256 * 256 * 2);
  bf16_t* wp3 = (bf16_t*)take((size_t)256 * 2304 * 2);
  float* sc_p1 = (float*)take(128 * 4); float* bi_p1 = (float*)take(128 * 4);
  float* sc_p2 = (float*)take(128 * 4); float* bi_p2 = (float*)take(128 * 4);
  float* sc_c1 = (float*)take(256 * 4); float* bi_c1 = (float*)take(256 * 4);
  float* sc_c2 = (float*)take(256 * 4); float* bi_c2 = (float*)take(256 * 4);
  float* sc_p3 = (float*)take(256 * 4); float* bi_p3 = (float*)take(256 * 4);
  bf16_t* rbuf = xbf;   // r overlays x_bf16 (x dead after c2)

  // ---- prep: layout conversion, weight packing, BN folding ----
  cvt_nchw_to_nhwc_kernel<<<dim3(HWX / 32, 256 / 32, NBATCH), 256, 0, stream>>>(x, xbf);
  pack_w_kernel<<<(128 * 2304 + 255) / 256, 256, 0, stream>>>(w_p1, wp1, 128, 256, 3);
  pack_w_kernel<<<(128 * 2304 + 255) / 256, 256, 0, stream>>>(w_p2, wp2, 128, 256, 3);
  pack_w_kernel<<<(256 * 1152 + 255) / 256, 256, 0, stream>>>(w_c1, wc1, 256, 128, 3);
  pack_w_kernel<<<(256 * 256  + 255) / 256, 256, 0, stream>>>(w_c2, wc2, 256, 256, 1);
  pack_w_kernel<<<(256 * 2304 + 255) / 256, 256, 0, stream>>>(w_p3, wp3, 256, 256, 3);
  bn_prep_kernel<<<1, 256, 0, stream>>>((const float*)d_in[2],  (const float*)d_in[3],
                                        (const float*)d_in[4],  (const float*)d_in[5],  sc_p1, bi_p1, 128);
  bn_prep_kernel<<<1, 256, 0, stream>>>((const float*)d_in[7],  (const float*)d_in[8],
                                        (const float*)d_in[9],  (const float*)d_in[10], sc_p2, bi_p2, 128);
  bn_prep_kernel<<<1, 256, 0, stream>>>((const float*)d_in[12], (const float*)d_in[13],
                                        (const float*)d_in[14], (const float*)d_in[15], sc_c1, bi_c1, 256);
  bn_prep_kernel<<<1, 256, 0, stream>>>((const float*)d_in[17], (const float*)d_in[18],
                                        (const float*)d_in[19], (const float*)d_in[20], sc_c2, bi_c2, 256);
  bn_prep_kernel<<<1, 256, 0, stream>>>((const float*)d_in[22], (const float*)d_in[23],
                                        (const float*)d_in[24], (const float*)d_in[25], sc_p3, bi_p3, 256);

  const dim3 gridA(HWX / 64, 1, NBATCH);   // block covers full channel range

  // p1 = relu(bn(conv3x3(x)))   [256->128]  NTJ=2 -> block N = 128
  conv_wmma_kernel<256, 128, 3, 2, true,  false, false>
      <<<gridA, 256, 0, stream>>>(xbf, wp1, sc_p1, bi_p1, nullptr, p1, nullptr);
  // p2 = relu(bn(conv3x3(x)))   [256->128]
  conv_wmma_kernel<256, 128, 3, 2, true,  false, false>
      <<<gridA, 256, 0, stream>>>(xbf, wp2, sc_p2, bi_p2, nullptr, p2, nullptr);

  // corner pooling: cp1 = revcummax_H(p1) in place; p2 <- revcummax_W(p2) + cp1
  scan_h_kernel<<<(NBATCH * HWX / IMG_H), 256, 0, stream>>>(p1);          // 512 blocks
  scan_w_add_kernel<<<(NBATCH * HWX / IMG_W), 256, 0, stream>>>(p2, p1);  // p2 == s

  // bn2 = bn(conv1x1(x))  [256->256] -> d_out (NCHW f32 scratch)  NTJ=4 -> N=256
  conv_wmma_kernel<256, 256, 1, 4, false, false, true>
      <<<gridA, 256, 0, stream>>>(xbf, wc2, sc_c2, bi_c2, nullptr, nullptr, out);
  // r = relu(bn(conv3x3(s)) + bn2)  [128->256] -> NHWC bf16 (overlays xbf)
  conv_wmma_kernel<128, 256, 3, 4, true,  true,  false>
      <<<gridA, 256, 0, stream>>>(p2, wc1, sc_c1, bi_c1, out, rbuf, nullptr);
  // out = relu(bn(conv3x3(r)))  [256->256] -> d_out (NCHW f32, final)
  conv_wmma_kernel<256, 256, 3, 4, true,  false, true>
      <<<gridA, 256, 0, stream>>>(rbuf, wp3, sc_p3, bi_p3, nullptr, nullptr, out);
}
